// GatedGraphConv_81157702025491
// MI455X (gfx1250) — matrix operand order repaired and checked by reading, not compiled, not run on hardware
//
#include <hip/hip_runtime.h>
#include <hip/hip_bf16.h>
#include <math.h>
#include <stdint.h>

// ---------------------------------------------------------------------------
// GatedGraphConv on MI455X (gfx1250, wave32).
//  - GEMMs via v_wmma_f32_16x16x32_f16, one 16x16 D-tile per wave, K=128.
//  - A-tiles staged block-wide into LDS via global_load_async_to_lds_b128
//    (ASYNCcnt path), consumed via ds_load_b128 by all 8 waves.
//  - GRU fully fused: 6 WMMA accumulators per wave, gates in-register.
//  - Edge gather/scatter: f16 message rows + global_atomic_add_f32, with
//    global_prefetch_b8 on the next edge row.
// ---------------------------------------------------------------------------

typedef _Float16 half4   __attribute__((ext_vector_type(4)));
typedef _Float16 half8   __attribute__((ext_vector_type(8)));
typedef _Float16 v16h    __attribute__((ext_vector_type(16)));
typedef float    v8f     __attribute__((ext_vector_type(8)));
typedef float    float4v __attribute__((ext_vector_type(4)));

#define N_NODES 100000
#define E_EDGES 1600000
#define D_IN    64
#define DF      128
#define NTYPES  4
#define NSTEPS  6

// ---------------- gfx1250 async copy to LDS (inline asm, portable) ---------

__device__ __forceinline__ void async_copy_16B(const void* gsrc, void* lds_dst) {
    uint64_t ga = (uint64_t)(uintptr_t)gsrc;
    uint32_t la = (uint32_t)(uintptr_t)lds_dst;   // generic->LDS offset (addr[31:0])
    asm volatile("global_load_async_to_lds_b128 %0, %1, off"
                 :: "v"(la), "v"(ga) : "memory");
}

__device__ __forceinline__ void wait_async0() {
    asm volatile("s_wait_asynccnt 0x0" ::: "memory");
}

// ---------------- WMMA operand loaders (ISA 7.12.2 layouts) ----------------

// A-matrix 16x32 f16 fragment from an LDS-resident 16x128 row-major tile.
// lane holds row (lane&15); K chunks [kb,kb+8) and [kb+16,kb+24),
// kb = k0 + 8*(lane>=16).
__device__ __forceinline__ v16h load_a_lds(const _Float16* __restrict__ s,
                                           int k0, int lane) {
    int kb = k0 + ((lane & 16) ? 8 : 0);
    const _Float16* p = s + (lane & 15) * DF + kb;
    half8 lo = *(const half8*)(p);
    half8 hi = *(const half8*)(p + 16);
    v16h r;
#pragma unroll
    for (int i = 0; i < 8; ++i) { r[i] = lo[i]; r[i + 8] = hi[i]; }
    return r;
}

// B-matrix 32x16 f16: lane holds column (lane&15); K chunks [kb,kb+8) and
// [kb+16,kb+24), kb = k0 + 8*(lane>=16).  W is row-major [out, 128]; B[k,n] = W[col, k].
__device__ __forceinline__ v16h load_b_f16(const _Float16* __restrict__ W,
                                           int col, int k0, int lane) {
    int kb = k0 + ((lane & 16) ? 8 : 0);
    const _Float16* p = W + (size_t)col * DF + kb;
    half8 lo = *(const half8*)(p);
    half8 hi = *(const half8*)(p + 16);
    v16h r;
#pragma unroll
    for (int i = 0; i < 8; ++i) { r[i] = lo[i]; r[i + 8] = hi[i]; }
    return r;
}

__device__ __forceinline__ float sigmoidf_(float x) {
    return 1.0f / (1.0f + __expf(-x));
}

// ---------------- utility kernels ----------------

__global__ void k_zero_f32(float* __restrict__ p, int n4) {
    int i = blockIdx.x * blockDim.x + threadIdx.x;
    float4v z = {0.f, 0.f, 0.f, 0.f};
    for (; i < n4; i += gridDim.x * blockDim.x)
        ((float4v*)p)[i] = z;
}

__global__ void k_cvt_f32_to_f16(const float* __restrict__ in,
                                 _Float16* __restrict__ out, int n) {
    int i = blockIdx.x * blockDim.x + threadIdx.x;
    for (; i < n; i += gridDim.x * blockDim.x)
        out[i] = (_Float16)in[i];
}

// feat = pad(x, [N,64] -> [N,128])
__global__ void k_init_feat(const float* __restrict__ x,
                            float* __restrict__ feat) {
    int i = blockIdx.x * blockDim.x + threadIdx.x;
    int n = N_NODES * DF;
    for (; i < n; i += gridDim.x * blockDim.x) {
        int row = i >> 7, c = i & (DF - 1);
        feat[i] = (c < D_IN) ? x[row * D_IN + c] : 0.0f;
    }
}

// ---------------- per-type transform: H16[t] = feat @ W_lin[t]^T + b_lin[t] --

__global__ void __launch_bounds__(256)
k_type_gemm(const _Float16* __restrict__ feat16,
            const _Float16* __restrict__ wlin16,
            const float* __restrict__ b_lin,
            _Float16* __restrict__ H16) {
    __shared__ __align__(16) _Float16 sA[16 * DF];   // 4KB A-tile

    int lane  = threadIdx.x & 31;
    int wv    = threadIdx.x >> 5;          // 8 waves -> 8 channel tiles
    int mbase = blockIdx.x * 16;           // node tile (6250 exact tiles)
    int t     = blockIdx.y;                // edge type
    int nbase = wv * 16;
    const _Float16* W = wlin16 + (size_t)t * DF * DF;

    // stage 16x128 A-tile into LDS: 256 threads x 16B async copies
    async_copy_16B(feat16 + (size_t)mbase * DF + threadIdx.x * 8,
                   &sA[threadIdx.x * 8]);
    wait_async0();
    __syncthreads();

    int bcol = nbase + (lane & 15);
    v8f acc = {};
#pragma unroll
    for (int k0 = 0; k0 < DF; k0 += 32) {
        v16h A = load_a_lds(sA, k0, lane);
        v16h B = load_b_f16(W, bcol, k0, lane);
        acc = __builtin_amdgcn_wmma_f32_16x16x32_f16(
            false, A, false, B, (short)0, acc, false, false);
    }
    float bias = b_lin[t * DF + bcol];
    _Float16* outp = H16 + (size_t)t * N_NODES * DF;
    int rbase = (lane & 16) ? 8 : 0;       // D layout: vgpr r -> row r(+8)
#pragma unroll
    for (int r = 0; r < 8; ++r) {
        int m = mbase + rbase + r;
        outp[(size_t)m * DF + bcol] = (_Float16)(acc[r] + bias);
    }
}

// ---------------- edge gather + scatter-add -------------------------------
// one wave per edge; lane covers 4 channels (32*4 = 128)

__global__ void __launch_bounds__(256)
k_edge_scatter(const int* __restrict__ src, const int* __restrict__ dst,
               const int* __restrict__ et, const _Float16* __restrict__ H16,
               float* __restrict__ a32) {
    int lane   = threadIdx.x & 31;
    int wv     = threadIdx.x >> 5;
    int stride = gridDim.x * 8;
    for (int e = blockIdx.x * 8 + wv; e < E_EDGES; e += stride) {
        int s = src[e], d = dst[e], t = et[e];
        const _Float16* hp =
            H16 + ((size_t)t * N_NODES + s) * DF + lane * 4;
        half4 v = *(const half4*)hp;

        int en = e + stride;               // prefetch next edge's source row
        if (en < E_EDGES) {
            int sn = src[en], tn = et[en];
            __builtin_prefetch(
                H16 + ((size_t)tn * N_NODES + sn) * DF + lane * 4, 0, 0);
        }
        float* ap = a32 + (size_t)d * DF + lane * 4;
        atomicAdd(ap + 0, (float)v[0]);
        atomicAdd(ap + 1, (float)v[1]);
        atomicAdd(ap + 2, (float)v[2]);
        atomicAdd(ap + 3, (float)v[3]);
    }
}

// ---------------- fused GRU cell ------------------------------------------
// gi = a @ W_ih^T + b_ih ; gh = h @ W_hh^T + b_hh ; gates in-register.
// Each wave: 16-node x 16-channel tile, 6 accumulators (r/z/n for gi and gh).

__global__ void __launch_bounds__(256)
k_gru_fused(const _Float16* __restrict__ a16,
            const _Float16* __restrict__ h16,
            const float* __restrict__ feat32,
            const _Float16* __restrict__ wih16,
            const _Float16* __restrict__ whh16,
            const float* __restrict__ b_ih,
            const float* __restrict__ b_hh,
            float* __restrict__ out) {
    __shared__ __align__(16) _Float16 sAa[16 * DF];  // 4KB a-tile
    __shared__ __align__(16) _Float16 sAh[16 * DF];  // 4KB h-tile

    int lane  = threadIdx.x & 31;
    int wv    = threadIdx.x >> 5;
    int mbase = blockIdx.x * 16;
    int nbase = wv * 16;
    int c     = nbase + (lane & 15);       // output channel for this lane

    size_t tilebase = (size_t)mbase * DF + threadIdx.x * 8;
    async_copy_16B(a16 + tilebase, &sAa[threadIdx.x * 8]);
    async_copy_16B(h16 + tilebase, &sAh[threadIdx.x * 8]);
    wait_async0();
    __syncthreads();

    v8f ir = {}, iz = {}, in_ = {}, hr = {}, hz = {}, hn = {};
#pragma unroll
    for (int k0 = 0; k0 < DF; k0 += 32) {
        v16h Aa = load_a_lds(sAa, k0, lane);
        v16h Ah = load_a_lds(sAh, k0, lane);
        v16h B;
        B  = load_b_f16(wih16, c,           k0, lane);
        ir  = __builtin_amdgcn_wmma_f32_16x16x32_f16(false, Aa, false, B, (short)0, ir,  false, false);
        B  = load_b_f16(wih16, c + DF,      k0, lane);
        iz  = __builtin_amdgcn_wmma_f32_16x16x32_f16(false, Aa, false, B, (short)0, iz,  false, false);
        B  = load_b_f16(wih16, c + 2 * DF,  k0, lane);
        in_ = __builtin_amdgcn_wmma_f32_16x16x32_f16(false, Aa, false, B, (short)0, in_, false, false);
        B  = load_b_f16(whh16, c,           k0, lane);
        hr  = __builtin_amdgcn_wmma_f32_16x16x32_f16(false, Ah, false, B, (short)0, hr,  false, false);
        B  = load_b_f16(whh16, c + DF,      k0, lane);
        hz  = __builtin_amdgcn_wmma_f32_16x16x32_f16(false, Ah, false, B, (short)0, hz,  false, false);
        B  = load_b_f16(whh16, c + 2 * DF,  k0, lane);
        hn  = __builtin_amdgcn_wmma_f32_16x16x32_f16(false, Ah, false, B, (short)0, hn,  false, false);
    }

    float bir = b_ih[c], biz = b_ih[c + DF], bin = b_ih[c + 2 * DF];
    float bhr = b_hh[c], bhz = b_hh[c + DF], bhn = b_hh[c + 2 * DF];

    int rbase = (lane & 16) ? 8 : 0;
#pragma unroll
    for (int r = 0; r < 8; ++r) {
        int m   = mbase + rbase + r;
        float h = feat32[(size_t)m * DF + c];
        float rg = sigmoidf_((ir[r] + bir) + (hr[r] + bhr));
        float zg = sigmoidf_((iz[r] + biz) + (hz[r] + bhz));
        float ng = tanhf((in_[r] + bin) + rg * (hn[r] + bhn));
        out[(size_t)m * DF + c] = (1.0f - zg) * ng + zg * h;
    }
}

// ---------------- host driver ---------------------------------------------

extern "C" void kernel_launch(void* const* d_in, const int* in_sizes, int n_in,
                              void* d_out, int out_size, void* d_ws, size_t ws_size,
                              hipStream_t stream) {
    const float* x        = (const float*)d_in[0];
    const int*   eidx     = (const int*)d_in[1];   // [2, E] flat
    const int*   etypes   = (const int*)d_in[2];
    const float* W_lin    = (const float*)d_in[3]; // [T,128,128]
    const float* b_lin    = (const float*)d_in[4]; // [T,128]
    const float* W_ih     = (const float*)d_in[5]; // [384,128]
    const float* W_hh     = (const float*)d_in[6]; // [384,128]
    const float* b_ih     = (const float*)d_in[7]; // [384]
    const float* b_hh     = (const float*)d_in[8]; // [384]
    float*       out      = (float*)d_out;

    const int* src = eidx;
    const int* dst = eidx + E_EDGES;

    // workspace carve-up (256B aligned)
    char* ws = (char*)d_ws;
    size_t off = 0;
    auto carve = [&](size_t bytes) {
        char* p = ws + off;
        off = (off + bytes + 255) & ~(size_t)255;
        return p;
    };
    float*     feat32 = (float*)    carve((size_t)N_NODES * DF * 4);
    _Float16*  feat16 = (_Float16*) carve((size_t)N_NODES * DF * 2);
    float*     a32    = (float*)    carve((size_t)N_NODES * DF * 4);
    _Float16*  a16    = (_Float16*) carve((size_t)N_NODES * DF * 2);
    _Float16*  H16    = (_Float16*) carve((size_t)NTYPES * N_NODES * DF * 2);
    _Float16*  wlin16 = (_Float16*) carve((size_t)NTYPES * DF * DF * 2);
    _Float16*  wih16  = (_Float16*) carve((size_t)3 * DF * DF * 2);
    _Float16*  whh16  = (_Float16*) carve((size_t)3 * DF * DF * 2);
    (void)ws_size; (void)n_in; (void)in_sizes; (void)out_size;

    const int THREADS = 256;

    // one-time: convert weights to f16, build padded feat
    k_cvt_f32_to_f16<<<64, THREADS, 0, stream>>>(W_lin, wlin16, NTYPES * DF * DF);
    k_cvt_f32_to_f16<<<64, THREADS, 0, stream>>>(W_ih,  wih16,  3 * DF * DF);
    k_cvt_f32_to_f16<<<64, THREADS, 0, stream>>>(W_hh,  whh16,  3 * DF * DF);
    k_init_feat<<<2048, THREADS, 0, stream>>>(x, feat32);

    const int mtiles = N_NODES / 16;   // 6250, exact

    for (int step = 0; step < NSTEPS; ++step) {
        // a = 0
        k_zero_f32<<<2048, THREADS, 0, stream>>>(a32, N_NODES * DF / 4);
        // feat16 = f16(feat32)
        k_cvt_f32_to_f16<<<2048, THREADS, 0, stream>>>(feat32, feat16,
                                                       N_NODES * DF);
        // H16[t] = feat @ W_lin[t]^T + b_lin[t]
        k_type_gemm<<<dim3(mtiles, NTYPES), THREADS, 0, stream>>>(
            feat16, wlin16, b_lin, H16);
        // a[dst] += H16[etype, src]
        k_edge_scatter<<<8192, THREADS, 0, stream>>>(src, dst, etypes, H16, a32);
        // a16 = f16(a)
        k_cvt_f32_to_f16<<<2048, THREADS, 0, stream>>>(a32, a16, N_NODES * DF);
        // feat = GRU(a, feat)  (in-place safe; last step -> d_out)
        float* dst_feat = (step == NSTEPS - 1) ? out : feat32;
        k_gru_fused<<<mtiles, THREADS, 0, stream>>>(
            a16, feat16, feat32, wih16, whh16, b_ih, b_hh, dst_feat);
    }
}
